// Embeddings_2894807958250
// MI455X (gfx1250) — matrix operand order repaired and checked by reading, not compiled, not run on hardware
//
#include <hip/hip_runtime.h>
#include <hip/hip_bf16.h>
#include <cstdint>

// ---------------------------------------------------------------------------
// Problem constants
// ---------------------------------------------------------------------------
constexpr int M_ROWS = 4 * 256;     // B*S = 1024
constexpr int D_DIM  = 768;         // K
constexpr int V_DIM  = 50257;       // N (vocab)
constexpr float INV_T = 10.0f;      // 1 / 0.1

// Output layout (flat float, in tuple return order)
constexpr size_t OUT0 = 0;                                  // pred_embeds_st  (M,768)
constexpr size_t OUT1 = OUT0 + (size_t)M_ROWS * D_DIM;      // pred_embeds     (M,768)
constexpr size_t OUT2 = OUT1 + (size_t)M_ROWS * D_DIM;      // predictions     (M,)
constexpr size_t OUT3 = OUT2 + (size_t)M_ROWS;              // pred_probs      (M,V) one-hot
constexpr size_t OUT4 = OUT3 + (size_t)M_ROWS * V_DIM;      // softmax probs   (M,V)

// Workspace layout (bytes)
constexpr size_t OFF_WBF  = 0;                                   // W as bf16 (V*768)
constexpr size_t SZ_WBF   = (size_t)V_DIM * D_DIM * 2;           // 77,194,752 (256-aligned)
constexpr size_t OFF_ABF  = OFF_WBF + SZ_WBF;                    // P as bf16 (M*768)
constexpr size_t SZ_ABF   = (size_t)M_ROWS * D_DIM * 2;
constexpr size_t OFF_TSQ  = OFF_ABF + SZ_ABF;                    // t_sq (V floats)
constexpr size_t SZ_TSQ   = (((size_t)V_DIM * 4) + 255) & ~(size_t)255;
constexpr size_t OFF_RMAX = OFF_TSQ + SZ_TSQ;                    // row max (M floats)
constexpr size_t OFF_RSUM = OFF_RMAX + 4096;                     // row sumexp
constexpr size_t OFF_PRED = OFF_RSUM + 4096;                     // argmax (M ints)

// ---------------------------------------------------------------------------
// Types for WMMA / async copies
// ---------------------------------------------------------------------------
typedef __attribute__((ext_vector_type(16))) __bf16        v16bf;
typedef __attribute__((ext_vector_type(8)))  float         v8f;
typedef __attribute__((ext_vector_type(8)))  unsigned int  v8u;
typedef int v4i __attribute__((vector_size(16)));   // matches builtin param type

__device__ __forceinline__ unsigned short f2bf(float f) {
    unsigned int u = __float_as_uint(f);
    u += 0x7FFFu + ((u >> 16) & 1u);          // round-to-nearest-even
    return (unsigned short)(u >> 16);
}

// ---------------------------------------------------------------------------
// gfx1250 async global->LDS copy (ASYNCcnt path), with synchronous fallback.
// ---------------------------------------------------------------------------
#if defined(__gfx1250__) && __has_builtin(__builtin_amdgcn_global_load_async_to_lds_b128)
#define USE_ASYNC_LDS 1
#else
#define USE_ASYNC_LDS 0
#endif

__device__ __forceinline__ void copy16_to_lds(const unsigned short* __restrict__ g,
                                              unsigned short* l) {
#if USE_ASYNC_LDS
    // global src pointer (AS1 = __device__) and LDS dst pointer (AS3, 32-bit offset)
    __attribute__((address_space(1))) v4i* gp =
        (__attribute__((address_space(1))) v4i*)(uintptr_t)(const void*)g;
    __attribute__((address_space(3))) v4i* lp =
        (__attribute__((address_space(3))) v4i*)(uintptr_t)
            (unsigned int)(uintptr_t)(void*)l;
    __builtin_amdgcn_global_load_async_to_lds_b128(gp, lp, 0, 0);
#else
    *(uint4*)l = *(const uint4*)g;
#endif
}

__device__ __forceinline__ void copy_wait() {
#if USE_ASYNC_LDS
#if __has_builtin(__builtin_amdgcn_s_wait_asynccnt)
    __builtin_amdgcn_s_wait_asynccnt(0);
#else
    asm volatile("s_wait_asynccnt 0x0" ::: "memory");
#endif
#endif
}

// ---------------------------------------------------------------------------
// Kernel 1a: weights -> bf16, plus t_sq[v] = sum_d W[v,d]^2  (fp32 source)
// ---------------------------------------------------------------------------
__global__ __launch_bounds__(256) void prep_w_kernel(const float* __restrict__ W,
                                                     unsigned short* __restrict__ Wbf,
                                                     float* __restrict__ tsq) {
    __shared__ float red[256];
    const int row = blockIdx.x;
    const int tid = threadIdx.x;
    float acc = 0.0f;
#pragma unroll
    for (int i = 0; i < 3; ++i) {
        const int d = tid + i * 256;                     // 768 = 3*256
        const float f = W[(size_t)row * D_DIM + d];
        acc += f * f;
        Wbf[(size_t)row * D_DIM + d] = f2bf(f);
    }
    red[tid] = acc;
    __syncthreads();
    for (int off = 128; off > 0; off >>= 1) {
        if (tid < off) red[tid] += red[tid + off];
        __syncthreads();
    }
    if (tid == 0) tsq[row] = red[0];
}

// ---------------------------------------------------------------------------
// Kernel 1b: pred_embeds -> bf16
// ---------------------------------------------------------------------------
__global__ __launch_bounds__(256) void prep_p_kernel(const float* __restrict__ P,
                                                     unsigned short* __restrict__ Abf) {
    const int row = blockIdx.x;
    const int tid = threadIdx.x;
#pragma unroll
    for (int i = 0; i < 3; ++i) {
        const int d = tid + i * 256;
        Abf[(size_t)row * D_DIM + d] = f2bf(P[(size_t)row * D_DIM + d]);
    }
}

// ---------------------------------------------------------------------------
// Kernel 2: WMMA bf16 GEMM, double-buffered LDS with async global->LDS copies.
// Block tile 128(M) x 128(N), K-step 32, 8 waves -> each wave 32x64.
// Writes staged scores (2*dot - t_sq)*10 into the softmax output slot (out4).
// ---------------------------------------------------------------------------
__global__ __launch_bounds__(256) void gemm_scores_kernel(const unsigned short* __restrict__ Abf,
                                                          const unsigned short* __restrict__ Wbf,
                                                          const float* __restrict__ tsq,
                                                          float* __restrict__ scores) {
    // 80B row stride: 16B aligned rows, conflict-friendly strided reads
    __shared__ unsigned short As[2][128][40];
    __shared__ unsigned short Bs[2][128][40];   // Bs[buf][n][k] = W[n0+n][k0+k]

    const int tid  = threadIdx.x;
    const int n0   = blockIdx.x * 128;
    const int m0   = blockIdx.y * 128;
    const int wave = tid >> 5;
    const int lane = tid & 31;
    const int hi   = lane >> 4;      // half-wave (0/1)
    const int l16  = lane & 15;
    const int wm   = (wave & 3) * 32;    // wave's M offset in tile
    const int wn   = (wave >> 2) * 64;   // wave's N offset in tile

    // Each thread stages 2 x 16B per tile: chunk = tid and tid+256 (of 512).
    const int r0 = tid >> 2,          c0 = (tid & 3) * 8;
    const int r1 = (tid + 256) >> 2,  c1 = (tid & 3) * 8;   // (tid+256)&3 == tid&3
    // Clamp B rows in-bounds (uniform, no EXEC divergence); clamped rows only
    // feed accumulator columns that the epilogue never stores.
    const int vb0 = (n0 + r0 < V_DIM) ? (n0 + r0) : (V_DIM - 1);
    const int vb1 = (n0 + r1 < V_DIM) ? (n0 + r1) : (V_DIM - 1);
    const unsigned short* Ag0 = Abf + (size_t)(m0 + r0) * D_DIM + c0;
    const unsigned short* Ag1 = Abf + (size_t)(m0 + r1) * D_DIM + c1;
    const unsigned short* Bg0 = Wbf + (size_t)vb0 * D_DIM + c0;
    const unsigned short* Bg1 = Wbf + (size_t)vb1 * D_DIM + c1;

    auto stage = [&](int buf, int k0) {
        copy16_to_lds(Ag0 + k0, &As[buf][r0][c0]);
        copy16_to_lds(Ag1 + k0, &As[buf][r1][c1]);
        copy16_to_lds(Bg0 + k0, &Bs[buf][r0][c0]);
        copy16_to_lds(Bg1 + k0, &Bs[buf][r1][c1]);
    };

    stage(0, 0);
    copy_wait();
    __syncthreads();

    v8f acc[2][4] = {};

    constexpr int KSTEPS = D_DIM / 32;   // 24
    for (int kt = 0; kt < KSTEPS; ++kt) {
        const int cur = kt & 1;
        if (kt + 1 < KSTEPS)
            stage(cur ^ 1, (kt + 1) * 32);   // overlap next-tile DMA with math

        // ---- build fragments per ISA 7.12.2 layouts
        v16bf afrag[2];
#pragma unroll
        for (int mi = 0; mi < 2; ++mi) {
            v8u au;
            const int m = wm + mi * 16 + l16;
#pragma unroll
            for (int j = 0; j < 8; ++j) {
                const int kk = (j < 4) ? (hi * 8 + 2 * j) : (16 + hi * 8 + 2 * (j - 4));
                au[j] = *(const unsigned int*)&As[cur][m][kk];
            }
            afrag[mi] = __builtin_bit_cast(v16bf, au);
        }
        v16bf bfrag[4];
#pragma unroll
        for (int ni = 0; ni < 4; ++ni) {
            v8u bu;
            const int n = wn + ni * 16 + l16;
#pragma unroll
            for (int j = 0; j < 8; ++j) {
                const int kk = (j < 4) ? (hi * 8 + 2 * j) : (16 + hi * 8 + 2 * (j - 4));
                bu[j] = *(const unsigned int*)&Bs[cur][n][kk];
            }
            bfrag[ni] = __builtin_bit_cast(v16bf, bu);
        }

        // ---- 8 WMMAs per wave per K-step
#pragma unroll
        for (int mi = 0; mi < 2; ++mi)
#pragma unroll
            for (int ni = 0; ni < 4; ++ni)
                acc[mi][ni] = __builtin_amdgcn_wmma_f32_16x16x32_bf16(
                    false, afrag[mi], false, bfrag[ni],
                    (short)0, acc[mi][ni], false, false);

        if (kt + 1 < KSTEPS)
            copy_wait();      // my async writes for buf^1 have landed in LDS
        __syncthreads();      // all waves done reading buf[cur]; writes visible
    }

    // ---- epilogue: scores = (2*dot - t_sq[col]) * (1/T), staged into out4
#pragma unroll
    for (int ni = 0; ni < 4; ++ni) {
        const int col = n0 + wn + ni * 16 + l16;
        if (col >= V_DIM) continue;
        const float tq = tsq[col];
#pragma unroll
        for (int mi = 0; mi < 2; ++mi) {
#pragma unroll
            for (int r = 0; r < 8; ++r) {
                const int row = m0 + wm + mi * 16 + hi * 8 + r;
                scores[(size_t)row * V_DIM + col] =
                    (2.0f * acc[mi][ni][r] - tq) * INV_T;
            }
        }
    }
}

// ---------------------------------------------------------------------------
// Kernel 3: per-row online logsumexp + argmax (single pass over V).
// ---------------------------------------------------------------------------
__global__ __launch_bounds__(256) void row_reduce_kernel(const float* __restrict__ scores,
                                                         float* __restrict__ rowmax,
                                                         float* __restrict__ rowsum,
                                                         int* __restrict__ pred,
                                                         float* __restrict__ out2) {
    __shared__ float smx[256];
    __shared__ float ssm[256];
    __shared__ int   sid[256];
    const int row = blockIdx.x;
    const int tid = threadIdx.x;
    const float* srow = scores + (size_t)row * V_DIM;

    float mx = -3.4e38f, sm = 0.0f;
    int idx = 0;
    for (int v = tid; v < V_DIM; v += 256) {
        const float s = srow[v];
        if (s > mx) { sm = sm * __expf(mx - s) + 1.0f; mx = s; idx = v; }
        else        { sm += __expf(s - mx); }
    }
    smx[tid] = mx; ssm[tid] = sm; sid[tid] = idx;
    __syncthreads();
    for (int off = 128; off > 0; off >>= 1) {
        if (tid < off) {
            const float m2 = smx[tid + off], s2 = ssm[tid + off];
            const int   i2 = sid[tid + off];
            float m1 = smx[tid], s1 = ssm[tid];
            if (m2 > m1) { s1 = s1 * __expf(m1 - m2) + s2; m1 = m2; sid[tid] = i2; }
            else         { s1 = s1 + s2 * __expf(m2 - m1); }
            smx[tid] = m1; ssm[tid] = s1;
        }
        __syncthreads();
    }
    if (tid == 0) {
        rowmax[row] = smx[0];
        rowsum[row] = ssm[0];
        pred[row]   = sid[0];
        out2[row]   = (float)sid[0];   // predictions as float in the flat output
    }
}

// ---------------------------------------------------------------------------
// Kernel 4: normalize softmax in place (out4) and emit one-hot (out3).
// ---------------------------------------------------------------------------
__global__ __launch_bounds__(256) void finalize_kernel(float* __restrict__ probs,   // out4 (staged scores)
                                                       float* __restrict__ onehot,  // out3
                                                       const float* __restrict__ rowmax,
                                                       const float* __restrict__ rowsum,
                                                       const int* __restrict__ pred) {
    const int row = blockIdx.y;
    const int v   = blockIdx.x * 256 + threadIdx.x;
    if (v >= V_DIM) return;
    const size_t o = (size_t)row * V_DIM + v;
    const float p = __expf(probs[o] - rowmax[row]) / rowsum[row];
    probs[o]  = p;
    onehot[o] = (v == pred[row]) ? 1.0f : 0.0f;
}

// ---------------------------------------------------------------------------
// Kernel 5: out0 = embed_weight[pred], out1 = copy(pred_embeds).
// ---------------------------------------------------------------------------
__global__ __launch_bounds__(256) void gather_copy_kernel(const float* __restrict__ P,
                                                          const float* __restrict__ W,
                                                          const int* __restrict__ pred,
                                                          float* __restrict__ out0,
                                                          float* __restrict__ out1) {
    const int row = blockIdx.x;
    const int tid = threadIdx.x;
    const int pv  = pred[row];
#pragma unroll
    for (int i = 0; i < 3; ++i) {
        const int d = tid + i * 256;
        out1[(size_t)row * D_DIM + d] = P[(size_t)row * D_DIM + d];
        out0[(size_t)row * D_DIM + d] = W[(size_t)pv * D_DIM + d];
    }
}

// ---------------------------------------------------------------------------
// Host-side launcher
// ---------------------------------------------------------------------------
extern "C" void kernel_launch(void* const* d_in, const int* in_sizes, int n_in,
                              void* d_out, int out_size, void* d_ws, size_t ws_size,
                              hipStream_t stream) {
    const float* pred_embeds  = (const float*)d_in[0];   // (4,256,768)
    const float* embed_weight = (const float*)d_in[1];   // (50257,768)

    float* out  = (float*)d_out;
    float* out0 = out + OUT0;
    float* out1 = out + OUT1;
    float* out2 = out + OUT2;
    float* out3 = out + OUT3;
    float* out4 = out + OUT4;   // staged scores -> softmax probs

    char* ws = (char*)d_ws;
    unsigned short* Wbf = (unsigned short*)(ws + OFF_WBF);
    unsigned short* Abf = (unsigned short*)(ws + OFF_ABF);
    float* tsq    = (float*)(ws + OFF_TSQ);
    float* rowmax = (float*)(ws + OFF_RMAX);
    float* rowsum = (float*)(ws + OFF_RSUM);
    int*   predi  = (int*)  (ws + OFF_PRED);

    // 1) precision conversion + t_sq
    prep_w_kernel<<<V_DIM, 256, 0, stream>>>(embed_weight, Wbf, tsq);
    prep_p_kernel<<<M_ROWS, 256, 0, stream>>>(pred_embeds, Abf);

    // 2) WMMA GEMM -> staged shifted/scaled scores in out4
    dim3 ggrid((V_DIM + 127) / 128, M_ROWS / 128);
    gemm_scores_kernel<<<ggrid, 256, 0, stream>>>(Abf, Wbf, tsq, out4);

    // 3) per-row max/argmax/sumexp (single pass)
    row_reduce_kernel<<<M_ROWS, 256, 0, stream>>>(out4, rowmax, rowsum, predi, out2);

    // 4) softmax normalize in place + one-hot
    dim3 fgrid((V_DIM + 255) / 256, M_ROWS);
    finalize_kernel<<<fgrid, 256, 0, stream>>>(out4, out3, rowmax, rowsum, predi);

    // 5) straight-through outputs
    gather_copy_kernel<<<M_ROWS, 256, 0, stream>>>(pred_embeds, embed_weight, predi,
                                                   out0, out1);
}
